// Pair_Stack_8538394985036
// MI455X (gfx1250) — compile-verified, matrix operand order checked by reading, not run on hardware
//
#include <hip/hip_runtime.h>
#include <hip/hip_bf16.h>

// ---------------------------------------------------------------------------
// Pair stack (triangle row/col attention + transition) for MI455X (gfx1250).
// All GEMM-shaped math runs on v_wmma_f32_16x16x32_bf16 (wave32 WMMA).
// Round 3: zero-padded Q/K tiles in LDS (no per-fragment VALU padding),
// pair-bias fused into the WMMA C accumulator, 32-bit hot-path addressing.
// ---------------------------------------------------------------------------

typedef __attribute__((ext_vector_type(16))) __bf16 v16bf;
typedef __attribute__((ext_vector_type(4)))  __bf16 v4bf;
typedef __attribute__((ext_vector_type(8)))  float  v8f;

#define WMMA_BF16(a, b, c) \
  __builtin_amdgcn_wmma_f32_16x16x32_bf16(false, (a), false, (b), (short)0, (c), false, false)

static constexpr int Bb = 2, Ls = 192, Cc = 128, Hh = 8;
static constexpr int LL = Ls * Ls;
static constexpr int Mrows = Bb * Ls * Ls;          // 73728
static constexpr float EPSf = 1e-5f;

enum { F_RELU = 1, F_TRANSA = 2, F_TRANSOUT = 4, F_RES = 8, F_BIAS = 16 };

__device__ __forceinline__ __bf16 f2bf(float f) { return (__bf16)f; }

// row m = (b,p,q) -> memory row (b,q,p)  (compile-time L -> mul/shift math)
__device__ __forceinline__ int swap_pq(int m) {
  int bI = m / LL; int rm = m - bI * LL;
  int p = rm / Ls; int q = rm - p * Ls;
  return (bI * Ls + q) * Ls + p;
}

// ---------------------------------------------------------------------------
// LayerNorm: one wave32 per row of C=128 (4 floats per lane).
// ---------------------------------------------------------------------------
__global__ __launch_bounds__(256)
void ln_kernel(const float* __restrict__ x, const float* __restrict__ g,
               const float* __restrict__ b, float* __restrict__ out, int M)
{
  int wave = (blockIdx.x * blockDim.x + threadIdx.x) >> 5;
  int lane = threadIdx.x & 31;
  if (wave >= M) return;
  const float4* xp = (const float4*)(x + (size_t)wave * Cc);
  float4 v = xp[lane];
  float s  = v.x + v.y + v.z + v.w;
  float s2 = v.x * v.x + v.y * v.y + v.z * v.z + v.w * v.w;
  #pragma unroll
  for (int off = 16; off; off >>= 1) {
    s  += __shfl_xor(s,  off, 32);
    s2 += __shfl_xor(s2, off, 32);
  }
  float mean = s * (1.0f / Cc);
  float var  = s2 * (1.0f / Cc) - mean * mean;
  float inv  = rsqrtf(var + EPSf);
  float4 gg = ((const float4*)g)[lane];
  float4 bb = ((const float4*)b)[lane];
  float4 o;
  o.x = (v.x - mean) * inv * gg.x + bb.x;
  o.y = (v.y - mean) * inv * gg.y + bb.y;
  o.z = (v.z - mean) * inv * gg.z + bb.z;
  o.w = (v.w - mean) * inv * gg.w + bb.w;
  ((float4*)(out + (size_t)wave * Cc))[lane] = o;
}

// ---------------------------------------------------------------------------
// Generic WMMA GEMM: out[M,N] = op(A[M,K] @ W[K,N] (+bias) (+relu)) (+res)
// Block: 128 threads (4 waves), 64x64 tile, K staged in 64-chunks (bf16 LDS).
// FLAGS is compile-time; all hot-path offsets are 32-bit.
// M, N, K multiples of 64 (true for all uses here).
// ---------------------------------------------------------------------------
template<int FLAGS>
__global__ __launch_bounds__(128)
void gemm_kernel(const float* __restrict__ A, const float* __restrict__ W,
                 const float* __restrict__ bias, const float* __restrict__ res,
                 float* __restrict__ out, int M, int N, int K)
{
  __shared__ __bf16 As[64][64 + 8];   // [row][k]
  __shared__ __bf16 Bs[64][64 + 8];   // [n][k]  (transposed in LDS)
  const int tid  = threadIdx.x;
  const int wave = tid >> 5;
  const int lane = tid & 31;
  const int ln   = lane & 15;
  const int lg   = lane >> 4;         // lane group (0 or 1)
  const int m0   = blockIdx.x * 64;
  const int n0   = blockIdx.y * 64;
  const int ar   = tid >> 4;          // staging row  0..7 (step 8)
  const int ac   = (tid & 15) * 4;    // staging col  0..60 (float4)

  // Precompute gathered A row offsets (32-bit; max index 73728*512 < 2^31).
  int aoff[8];
  #pragma unroll
  for (int i = 0; i < 8; ++i) {
    int m = m0 + ar + i * 8;
    aoff[i] = ((FLAGS & F_TRANSA) ? swap_pq(m) : m) * K;
  }

  v8f acc[4] = {v8f{}, v8f{}, v8f{}, v8f{}};

  for (int k0 = 0; k0 < K; k0 += 64) {
    // Prefetch next chunk into cache while we stage/compute this one.
    if (k0 + 64 < K) {
      __builtin_prefetch(&A[aoff[0] + k0 + 64 + ac], 0, 1);
      __builtin_prefetch(&W[(k0 + 64 + ar) * N + n0 + ac], 0, 1);
    }
    // Stage A tile (64 rows x 64 k): float4 global read -> packed b64 LDS write
    #pragma unroll
    for (int i = 0; i < 8; ++i) {
      int r = ar + i * 8;
      float4 v = *(const float4*)(A + aoff[i] + k0 + ac);
      v4bf w = {f2bf(v.x), f2bf(v.y), f2bf(v.z), f2bf(v.w)};
      *(v4bf*)(&As[r][ac]) = w;
    }
    // Stage B tile (64 k x 64 n) transposed: Bs[n][k]
    #pragma unroll
    for (int i = 0; i < 8; ++i) {
      int r = ar + i * 8;                       // k-local
      float4 v = *(const float4*)(W + (k0 + r) * N + n0 + ac);
      Bs[ac + 0][r] = f2bf(v.x);
      Bs[ac + 1][r] = f2bf(v.y);
      Bs[ac + 2][r] = f2bf(v.z);
      Bs[ac + 3][r] = f2bf(v.w);
    }
    __syncthreads();

    #pragma unroll
    for (int ks = 0; ks < 2; ++ks) {
      const int kbase = ks * 32 + (lg << 3);
      v16bf af;
      const int arw = wave * 16 + ln;
      #pragma unroll
      for (int e = 0; e < 16; ++e) {
        int kk = kbase + (e < 8 ? e : 8 + e);
        af[e] = As[arw][kk];
      }
      #pragma unroll
      for (int ct = 0; ct < 4; ++ct) {
        v16bf bf_;
        const int bn = ct * 16 + ln;
        #pragma unroll
        for (int e = 0; e < 16; ++e) {
          int kk = kbase + (e < 8 ? e : 8 + e);
          bf_[e] = Bs[bn][kk];
        }
        acc[ct] = WMMA_BF16(af, bf_, acc[ct]);
      }
    }
    __syncthreads();
  }

  // Epilogue: D layout -> global (row = vgpr idx, col = lane; lg selects M+8)
  #pragma unroll
  for (int ct = 0; ct < 4; ++ct) {
    #pragma unroll
    for (int v = 0; v < 8; ++v) {
      int m = m0 + wave * 16 + (lg << 3) + v;
      int n = n0 + ct * 16 + ln;
      float val = acc[ct][v];
      if (FLAGS & F_BIAS) val += bias[n];
      if (FLAGS & F_RELU) val = fmaxf(val, 0.0f);
      int orow = (FLAGS & F_TRANSOUT) ? swap_pq(m) : m;
      int idx = orow * N + n;               // < 2^31 for all uses
      if (FLAGS & F_RES) val += res[idx];
      out[idx] = val;
    }
  }
}

// ---------------------------------------------------------------------------
// Pair bias: out[m, h] = xn[mrow, :] . Wb[:, h] + bb[h]   (N = 8, VALU)
// ---------------------------------------------------------------------------
template<int TRANSA>
__global__ __launch_bounds__(256)
void pair_bias_kernel(const float* __restrict__ xn, const float* __restrict__ Wb,
                      const float* __restrict__ bb, float* __restrict__ out, int M)
{
  int idx = blockIdx.x * blockDim.x + threadIdx.x;
  if (idx >= M * Hh) return;
  int m = idx >> 3, h = idx & 7;
  int mrow = TRANSA ? swap_pq(m) : m;
  const float* xp = xn + mrow * Cc;
  float acc = bb[h];
  #pragma unroll 4
  for (int c = 0; c < Cc; ++c) acc += xp[c] * Wb[c * Hh + h];
  out[m * Hh + h] = acc;
}

// ---------------------------------------------------------------------------
// Attention: one block per (head, row r, batch b). 128 threads = 4 waves.
// qkv feature layout: f = d*32 + k*8 + h  (k: 0=q 1=k 2=v 3=gate).
// Q/K staged zero-padded to K=32 (unconditional b128 fragment loads);
// pair bias preloaded into the WMMA C accumulator; exact softmax;
// P bounced through LDS into A-frag layout; O = P @ V in 6 WMMA K-steps;
// sigmoid(gate) fused at store. Writes attn_out[m, h*16+d].
// ---------------------------------------------------------------------------
__global__ __launch_bounds__(128)
void attn_kernel(const float* __restrict__ qkv, const float* __restrict__ pbias,
                 float* __restrict__ out)
{
  const int h = blockIdx.x;
  const int r = blockIdx.y;
  const int bI = blockIdx.z;
  const int tid = threadIdx.x;
  const int wave = tid >> 5;
  const int lane = tid & 31;
  const int ln = lane & 15;
  const int lg = lane >> 4;
  const int rowb = (bI * Ls + r) * Ls;     // + t -> row of qkv/out (32-bit)

  __shared__ __bf16 Qs[192][40];        // [t][k0..31], cols 16..31 zeroed
  __shared__ __bf16 Ks[192][40];        // [t][k0..31], cols 16..31 zeroed
  __shared__ __bf16 Vs[16][200];        // [d][t]
  __shared__ __bf16 Pl[4][16][192];     // per-wave probability strip

  for (int i = tid; i < 192 * 16; i += 128) {
    int t = i >> 4, d = i & 15;
    const float* base = qkv + (rowb + t) * 512 + d * 32 + h;
    Qs[t][d] = f2bf(base[0] * 0.25f);   // 1/sqrt(16)
    Qs[t][16 + d] = f2bf(0.0f);         // zero pad (K 16->32)
    Ks[t][d] = f2bf(base[8]);
    Ks[t][16 + d] = f2bf(0.0f);
    Vs[d][t] = f2bf(base[16]);
  }
  __syncthreads();

  for (int strip = wave; strip < 12; strip += 4) {
    const int i0 = strip * 16;

    // Q fragment (A layout; upper K half reads staged zeros)
    v16bf aq;
    #pragma unroll
    for (int e = 0; e < 16; ++e) {
      int kk = (lg << 3) + (e < 8 ? e : 8 + e);
      aq[e] = Qs[i0 + ln][kk];
    }

    // Scores: 12 tiles of 16x16; pair bias rides in as the C accumulator.
    v8f s[12];
    #pragma unroll
    for (int j = 0; j < 12; ++j) {
      v16bf bk;
      #pragma unroll
      for (int e = 0; e < 16; ++e) {
        int kk = (lg << 3) + (e < 8 ? e : 8 + e);
        bk[e] = Ks[j * 16 + ln][kk];
      }
      // bias element for (i = i0 + lg*8 + v, jcol = 16j + ln), D layout
      const int boff = ((bI * Ls + i0 + (lg << 3)) * Ls + j * 16 + ln) * Hh + h;
      v8f cb;
      #pragma unroll
      for (int v = 0; v < 8; ++v) cb[v] = pbias[boff + v * (Ls * Hh)];
      s[j] = WMMA_BF16(aq, bk, cb);
    }

    // Exact softmax per query row (row i lives at vgpr v across a 16-lane group)
    #pragma unroll
    for (int v = 0; v < 8; ++v) {
      float mx = -1e30f;
      #pragma unroll
      for (int j = 0; j < 12; ++j) mx = fmaxf(mx, s[j][v]);
      #pragma unroll
      for (int off = 8; off; off >>= 1) mx = fmaxf(mx, __shfl_xor(mx, off, 32));
      float sum = 0.0f;
      #pragma unroll
      for (int j = 0; j < 12; ++j) {
        float p = __expf(s[j][v] - mx);
        s[j][v] = p;
        sum += p;
      }
      #pragma unroll
      for (int off = 8; off; off >>= 1) sum += __shfl_xor(sum, off, 32);
      float inv = 1.0f / sum;
      #pragma unroll
      for (int j = 0; j < 12; ++j) s[j][v] *= inv;
    }

    // Spill P (D layout) to LDS, re-read in A-frag layout.
    #pragma unroll
    for (int j = 0; j < 12; ++j)
      #pragma unroll
      for (int v = 0; v < 8; ++v)
        Pl[wave][(lg << 3) + v][j * 16 + ln] = f2bf(s[j][v]);

    // O = P @ V : K = 192 in 6 chunks of 32
    v8f o = {};
    #pragma unroll
    for (int kc = 0; kc < 6; ++kc) {
      v16bf ap, bv;
      #pragma unroll
      for (int e = 0; e < 16; ++e) {
        int kk = kc * 32 + (lg << 3) + (e < 8 ? e : 8 + e);
        ap[e] = Pl[wave][ln][kk];
        bv[e] = Vs[ln][kk];
      }
      o = WMMA_BF16(ap, bv, o);
    }

    // Sigmoid gate + store (feature = h*16 + d)
    #pragma unroll
    for (int v = 0; v < 8; ++v) {
      int t = i0 + (lg << 3) + v;
      int d = ln;
      float gval = qkv[(rowb + t) * 512 + d * 32 + 24 + h];
      float val = o[v] * (1.0f / (1.0f + __expf(-gval)));
      out[(rowb + t) * Cc + h * 16 + d] = val;
    }
  }
}

// ---------------------------------------------------------------------------
// Host-side orchestration
// ---------------------------------------------------------------------------
extern "C" void kernel_launch(void* const* d_in, const int* in_sizes, int n_in,
                              void* d_out, int out_size, void* d_ws, size_t ws_size,
                              hipStream_t stream) {
  const float* x      = (const float*)d_in[0];
  const float* Wqkv_s = (const float*)d_in[1];
  const float* W0_s   = (const float*)d_in[2];
  const float* Wb_s   = (const float*)d_in[3];
  const float* bb_s   = (const float*)d_in[4];
  const float* Wqkv_e = (const float*)d_in[5];
  const float* W0_e   = (const float*)d_in[6];
  const float* Wb_e   = (const float*)d_in[7];
  const float* bb_e   = (const float*)d_in[8];
  const float* g1     = (const float*)d_in[9];
  const float* b1     = (const float*)d_in[10];
  const float* g2     = (const float*)d_in[11];
  const float* b2     = (const float*)d_in[12];
  const float* W1     = (const float*)d_in[13];
  const float* bf1    = (const float*)d_in[14];
  const float* W2     = (const float*)d_in[15];
  const float* bf2    = (const float*)d_in[16];
  float* outp = (float*)d_out;

  const int M = Mrows;                  // 73728
  char* ws = (char*)d_ws;
  size_t off = 0;
  auto carve = [&](size_t bytes) { char* p = ws + off; off += (bytes + 255) & ~size_t(255); return (float*)p; };
  float* xn   = carve((size_t)M * Cc * 4);      //  38 MB (reused as xn2)
  float* qkvb = carve((size_t)M * 512 * 4);     // 151 MB (reused as MLP hidden)
  float* pbias= carve((size_t)M * Hh * 4);      // 2.4 MB
  float* attn = carve((size_t)M * Cc * 4);      //  38 MB
  float* x1   = carve((size_t)M * Cc * 4);      //  38 MB
  (void)ws_size;

  dim3 lnGrid(M / 8), lnBlk(256);
  dim3 biasGrid((M * Hh + 255) / 256), biasBlk(256);
  dim3 gemmBlk(128);
  dim3 attnGrid(Hh, Ls, Bb), attnBlk(128);

  // ---- row-wise pass ----
  ln_kernel<<<lnGrid, lnBlk, 0, stream>>>(x, g1, b1, xn, M);
  pair_bias_kernel<0><<<biasGrid, biasBlk, 0, stream>>>(xn, Wb_s, bb_s, pbias, M);
  gemm_kernel<0><<<dim3(M / 64, 512 / 64), gemmBlk, 0, stream>>>(
      xn, Wqkv_s, nullptr, nullptr, qkvb, M, 512, Cc);
  attn_kernel<<<attnGrid, attnBlk, 0, stream>>>(qkvb, pbias, attn);
  gemm_kernel<F_RES><<<dim3(M / 64, Cc / 64), gemmBlk, 0, stream>>>(
      attn, W0_s, nullptr, xn, x1, M, Cc, Cc);

  // ---- column-wise pass (operates in transposed index space) ----
  ln_kernel<<<lnGrid, lnBlk, 0, stream>>>(x1, g2, b2, xn, M);     // xn := xn2
  pair_bias_kernel<1><<<biasGrid, biasBlk, 0, stream>>>(xn, Wb_e, bb_e, pbias, M);
  gemm_kernel<F_TRANSA><<<dim3(M / 64, 512 / 64), gemmBlk, 0, stream>>>(
      xn, Wqkv_e, nullptr, nullptr, qkvb, M, 512, Cc);
  attn_kernel<<<attnGrid, attnBlk, 0, stream>>>(qkvb, pbias, attn);
  gemm_kernel<F_RES | F_TRANSOUT><<<dim3(M / 64, Cc / 64), gemmBlk, 0, stream>>>(
      attn, W0_e, nullptr, x1, outp, M, Cc, Cc);                  // x2 -> d_out

  // ---- transition MLP: relu(x2 @ W1 + bf1) @ W2 + bf2 + x2 ----
  gemm_kernel<F_RELU | F_BIAS><<<dim3(M / 64, 512 / 64), gemmBlk, 0, stream>>>(
      outp, W1, bf1, nullptr, qkvb, M, 512, Cc);
  gemm_kernel<F_BIAS | F_RES><<<dim3(M / 64, Cc / 64), gemmBlk, 0, stream>>>(
      qkvb, W2, bf2, outp, outp, M, Cc, 512);
}
